// NavAssignNet_62113817035076
// MI455X (gfx1250) — matrix operand order repaired and checked by reading, not compiled, not run on hardware
//
#include <hip/hip_runtime.h>
#include <hip/hip_bf16.h>

// ---------------------------------------------------------------------------
// NavAssignNet forward for MI455X (gfx1250, wave32, WMMA).
// B=8, N=64, D=128, H=256, L=3. All GEMMs run through v_wmma_f32_16x16x32_f16
// with f32 accumulation; activations/weights staged in f16 (weights
// pre-transposed to [N][K] so B-fragments are contiguous 16B loads).
// Entire working set (~60MB) is L2-resident on MI455X (192MB L2).
// ---------------------------------------------------------------------------

#define B_ 8
#define N_ 64
#define D_ 128
#define H_ 256
#define L_ 3

typedef _Float16 v16h __attribute__((ext_vector_type(16)));
typedef _Float16 v8h  __attribute__((ext_vector_type(8)));
typedef float    v8f  __attribute__((ext_vector_type(8)));

// ---------------------------------------------------------------------------
// target_raw[b,j,d] = mean_i edge_raw[b,i,j,d]
__global__ void nav_mean_i_kernel(const float* __restrict__ edge,
                                  float* __restrict__ out) {
    int idx = blockIdx.x * blockDim.x + threadIdx.x;   // B*N*D = 65536
    if (idx >= B_ * N_ * D_) return;
    int d = idx % D_;
    int j = (idx / D_) % N_;
    int b = idx / (D_ * N_);
    const float* p = edge + ((size_t)b * N_ * N_ + j) * D_ + d;
    float s = 0.f;
    #pragma unroll 8
    for (int i = 0; i < N_; ++i) s += p[(size_t)i * N_ * D_];
    out[idx] = s * (1.0f / N_);
}

// f32 -> f16 elementwise
__global__ void nav_cvt16_kernel(const float* __restrict__ in,
                                 _Float16* __restrict__ out, int n) {
    int idx = blockIdx.x * blockDim.x + threadIdx.x;
    if (idx < n) out[idx] = (_Float16)in[idx];
}

// Pack W[K][N] (f32, row-major) -> out[N][K] (f16): transposed for WMMA B side
__global__ void nav_packwt_kernel(const float* __restrict__ W,
                                  _Float16* __restrict__ out, int K, int Ncols) {
    int idx = blockIdx.x * blockDim.x + threadIdx.x;
    if (idx >= K * Ncols) return;
    int k = idx % K;
    int n = idx / K;
    out[(size_t)n * K + k] = (_Float16)W[(size_t)k * Ncols + n];
}

// ---------------------------------------------------------------------------
// Generic WMMA GEMM: C[M,256] = A16[M,K] * Wt16[256,K]^T (+bias).
// Block = 256 threads = 8 waves; block tile 64x64; each wave owns 2 C tiles.
// Fragment layouts per CDNA5 ISA 7.12.2 (wave32):
//   A 16x32 f16 : lane l holds row (l&15); K chunks {8*hi..8*hi+7} and
//                 {16+8*hi..23+8*hi} where hi = l>>4  -> two 16B loads.
//   B 32x16 f16 : lane l holds col (l&15); K chunks {8*hi..8*hi+7} and
//                 {16+8*hi..23+8*hi} in pre-transposed Wt -> two 16B loads.
//   C 16x16 f32 : lane l holds col (l&15); vgpr r -> row r + 8*hi.
__global__ __launch_bounds__(256) void nav_gemm_wmma_kernel(
    const _Float16* __restrict__ A, const _Float16* __restrict__ Wt,
    const float* __restrict__ bias, float* __restrict__ outF,
    _Float16* __restrict__ outH, int M, int Nn, int K) {
    int bm   = blockIdx.x * 64;
    int bn   = blockIdx.y * 64;
    int wave = threadIdx.x >> 5;
    int lane = threadIdx.x & 31;
    int ct   = wave & 3;         // col tile 0..3
    int rt   = wave >> 2;        // row tile pair 0..1 (rows rt and rt+2)
    int n0   = bn + ct * 16;
    int m0   = bm + rt * 16;
    int lr   = lane & 15;
    int hi   = lane >> 4;

    const _Float16* Arow0 = A + (size_t)(m0 + lr) * K;
    const _Float16* Arow1 = Arow0 + (size_t)32 * K;
    const _Float16* Brow  = Wt + (size_t)(n0 + lr) * K;
    const int koffA = hi * 8;
    const int koffB = hi * 16;

    v8f acc0 = {};
    v8f acc1 = {};
    for (int k = 0; k < K; k += 32) {
        v8h a0lo = *(const v8h*)(Arow0 + k + koffA);
        v8h a0hi = *(const v8h*)(Arow0 + k + 16 + koffA);
        v8h a1lo = *(const v8h*)(Arow1 + k + koffA);
        v8h a1hi = *(const v8h*)(Arow1 + k + 16 + koffA);
        v16h b   = __builtin_shufflevector(*(const v8h*)(Brow + k + koffA),
                                           *(const v8h*)(Brow + k + 16 + koffA),
                                           0, 1, 2, 3, 4, 5, 6, 7,
                                           8, 9, 10, 11, 12, 13, 14, 15);
        v16h a0  = __builtin_shufflevector(a0lo, a0hi, 0, 1, 2, 3, 4, 5, 6, 7,
                                           8, 9, 10, 11, 12, 13, 14, 15);
        v16h a1  = __builtin_shufflevector(a1lo, a1hi, 0, 1, 2, 3, 4, 5, 6, 7,
                                           8, 9, 10, 11, 12, 13, 14, 15);
        acc0 = __builtin_amdgcn_wmma_f32_16x16x32_f16(false, a0, false, b,
                                                      (short)0, acc0, false, false);
        acc1 = __builtin_amdgcn_wmma_f32_16x16x32_f16(false, a1, false, b,
                                                      (short)0, acc1, false, false);
    }

    int col = n0 + lr;
    float bv = bias ? bias[col] : 0.0f;
    #pragma unroll
    for (int r = 0; r < 8; ++r) {
        int mr0 = m0 + r + hi * 8;
        int mr1 = mr0 + 32;
        float v0 = acc0[r] + bv;
        float v1 = acc1[r] + bv;
        if (outF) {
            outF[(size_t)mr0 * Nn + col] = v0;
            outF[(size_t)mr1 * Nn + col] = v1;
        }
        if (outH) {
            outH[(size_t)mr0 * Nn + col] = (_Float16)v0;
            outH[(size_t)mr1 * Nn + col] = (_Float16)v1;
        }
    }
}

// ---------------------------------------------------------------------------
// agg[b,x,h] = (1/N) * sum_y relu(small[b,y,h] + big[b*N*N + x*stepX + y*stepY, h] + bias[h])
//   r2t: x=j, y=i  -> stepX=1,  stepY=N   (mean over axis 1)
//   t2r: x=i, y=j  -> stepX=N,  stepY=1   (mean over axis 2)
__global__ void nav_agg_kernel(const float* __restrict__ small,
                               const float* __restrict__ big,
                               const float* __restrict__ bias,
                               _Float16* __restrict__ out16,
                               int stepX, int stepY) {
    int idx = blockIdx.x * blockDim.x + threadIdx.x;   // B*N*H = 131072
    if (idx >= B_ * N_ * H_) return;
    int h = idx & (H_ - 1);
    int x = (idx >> 8) & (N_ - 1);
    int b = idx >> 14;
    float bv = bias[h];
    const float* sm = small + ((size_t)b * N_) * H_ + h;
    size_t bigBase = ((size_t)b * N_ * N_ + (size_t)x * stepX) * H_ + h;
    float s = 0.f;
    for (int y = 0; y < N_; ++y) {
        float v = sm[(size_t)y * H_] + big[bigBase + (size_t)y * stepY * H_] + bv;
        s += fmaxf(v, 0.f);
    }
    out16[idx] = (_Float16)(s * (1.0f / N_));
}

// ---------------------------------------------------------------------------
// resid = LN(resid + u1 + u2 + bias; g, be); also write f16 copy.
__global__ __launch_bounds__(256) void nav_ln_kernel(
    float* __restrict__ resid, const float* __restrict__ u1,
    const float* __restrict__ u2, const float* __restrict__ bias,
    const float* __restrict__ g, const float* __restrict__ be,
    _Float16* __restrict__ out16) {
    __shared__ float red[256];
    int row = blockIdx.x;
    int h = threadIdx.x;
    size_t i = (size_t)row * H_ + h;
    float x = resid[i] + u1[i] + u2[i] + bias[h];

    red[h] = x;
    __syncthreads();
    for (int s = 128; s > 0; s >>= 1) {
        if (h < s) red[h] += red[h + s];
        __syncthreads();
    }
    float m = red[0] * (1.0f / H_);
    __syncthreads();

    float d = x - m;
    red[h] = d * d;
    __syncthreads();
    for (int s = 128; s > 0; s >>= 1) {
        if (h < s) red[h] += red[h + s];
        __syncthreads();
    }
    float var = red[0] * (1.0f / H_);

    float y = d * rsqrtf(var + 1e-5f) * g[h] + be[h];
    resid[i] = y;
    out16[i] = (_Float16)y;
}

// ---------------------------------------------------------------------------
// scores[b,i,j] = sum_h relu(rS[b,i,h] + tS[b,j,h] + bs1[h]) * Ws2[h] + bs2
__global__ __launch_bounds__(256) void nav_score_kernel(
    const float* __restrict__ rS, const float* __restrict__ tS,
    const float* __restrict__ bs1, const float* __restrict__ Ws2,
    const float* __restrict__ bs2, float* __restrict__ scores) {
    __shared__ float red[256];
    int e = blockIdx.x;                 // b*N*N + i*N + j
    int j = e & (N_ - 1);
    int i = (e >> 6) & (N_ - 1);
    int b = e >> 12;
    int h = threadIdx.x;
    float v = fmaxf(rS[((size_t)b * N_ + i) * H_ + h] +
                    tS[((size_t)b * N_ + j) * H_ + h] + bs1[h], 0.f) * Ws2[h];
    red[h] = v;
    __syncthreads();
    for (int s = 128; s > 0; s >>= 1) {
        if (h < s) red[h] += red[h + s];
        __syncthreads();
    }
    if (h == 0) scores[e] = red[0] + bs2[0];
}

// ---------------------------------------------------------------------------
// Sinkhorn: 10 iterations of row-LSE then col-LSE normalization, then exp.
__global__ __launch_bounds__(64) void nav_sinkhorn_kernel(
    const float* __restrict__ scores, float* __restrict__ P) {
    __shared__ float ls[N_][N_ + 1];
    int b = blockIdx.x;
    int t = threadIdx.x;
    for (int r = 0; r < N_; ++r)
        ls[r][t] = scores[((size_t)b * N_ + r) * N_ + t];
    __syncthreads();
    for (int it = 0; it < 10; ++it) {
        // rows (axis -1)
        float m = -__builtin_inff();
        for (int j = 0; j < N_; ++j) m = fmaxf(m, ls[t][j]);
        float s = 0.f;
        for (int j = 0; j < N_; ++j) s += expf(ls[t][j] - m);
        float lse = m + logf(s);
        for (int j = 0; j < N_; ++j) ls[t][j] -= lse;
        __syncthreads();
        // cols (axis -2)
        m = -__builtin_inff();
        for (int i = 0; i < N_; ++i) m = fmaxf(m, ls[i][t]);
        s = 0.f;
        for (int i = 0; i < N_; ++i) s += expf(ls[i][t] - m);
        lse = m + logf(s);
        for (int i = 0; i < N_; ++i) ls[i][t] -= lse;
        __syncthreads();
    }
    for (int r = 0; r < N_; ++r)
        P[((size_t)b * N_ + r) * N_ + t] = expf(ls[r][t]);
}

// ---------------------------------------------------------------------------
extern "C" void kernel_launch(void* const* d_in, const int* in_sizes, int n_in,
                              void* d_out, int out_size, void* d_ws, size_t ws_size,
                              hipStream_t stream) {
    const float* robot_raw = (const float*)d_in[0];
    const float* edge_raw  = (const float*)d_in[1];
    const float* Wp    = (const float*)d_in[2];
    const float* bp    = (const float*)d_in[3];
    const float* W_r2t = (const float*)d_in[4];
    const float* b_r2t = (const float*)d_in[5];
    const float* W_pt  = (const float*)d_in[6];
    const float* b_pt  = (const float*)d_in[7];
    const float* g_t   = (const float*)d_in[8];
    const float* be_t  = (const float*)d_in[9];
    const float* W_t2r = (const float*)d_in[10];
    const float* b_t2r = (const float*)d_in[11];
    const float* W_pr  = (const float*)d_in[12];
    const float* b_pr  = (const float*)d_in[13];
    const float* g_r   = (const float*)d_in[14];
    const float* be_r  = (const float*)d_in[15];
    const float* Ws1   = (const float*)d_in[16];
    const float* bs1   = (const float*)d_in[17];
    const float* Ws2   = (const float*)d_in[18];
    const float* bs2   = (const float*)d_in[19];
    float* out = (float*)d_out;                 // [P (32768) | scores (32768)]

    const int ME = B_ * N_ * N_;                // 32768 edge rows
    const int MS = B_ * N_;                     // 512 node rows

    // ---- workspace carve-out (256B aligned) --------------------------------
    char* wsp = (char*)d_ws;
    auto alloc = [&](size_t bytes) {
        char* p = wsp;
        wsp += (bytes + 255) & ~(size_t)255;
        return p;
    };
    float* target_raw = (float*)alloc((size_t)MS * D_ * 4);
    float* robot_h    = (float*)alloc((size_t)MS * H_ * 4);
    float* target_h   = (float*)alloc((size_t)MS * H_ * 4);
    float* u1         = (float*)alloc((size_t)MS * H_ * 4);
    float* u2         = (float*)alloc((size_t)MS * H_ * 4);
    float* eWb        = (float*)alloc((size_t)ME * H_ * 4);      // 33.5 MB
    _Float16* A_rob16  = (_Float16*)alloc((size_t)MS * D_ * 2);
    _Float16* A_tgt16  = (_Float16*)alloc((size_t)MS * D_ * 2);
    _Float16* A_edge16 = (_Float16*)alloc((size_t)ME * D_ * 2);  // 8.4 MB
    _Float16* edge_h16 = (_Float16*)alloc((size_t)ME * H_ * 2);  // 16.8 MB
    _Float16* robot_h16  = (_Float16*)alloc((size_t)MS * H_ * 2);
    _Float16* target_h16 = (_Float16*)alloc((size_t)MS * H_ * 2);
    _Float16* agg16      = (_Float16*)alloc((size_t)MS * H_ * 2);
    _Float16* WpT = (_Float16*)alloc((size_t)H_ * D_ * 2);
    _Float16* Wr2tA[L_], *Wr2tB[L_], *WptA[L_], *WptB[L_];
    _Float16* Wt2rA[L_], *Wt2rB[L_], *WprA[L_], *WprB[L_];
    for (int l = 0; l < L_; ++l) {
        Wr2tA[l] = (_Float16*)alloc((size_t)H_ * H_ * 2);
        Wr2tB[l] = (_Float16*)alloc((size_t)H_ * H_ * 2);
        WptA[l]  = (_Float16*)alloc((size_t)H_ * H_ * 2);
        WptB[l]  = (_Float16*)alloc((size_t)H_ * H_ * 2);
        Wt2rA[l] = (_Float16*)alloc((size_t)H_ * H_ * 2);
        Wt2rB[l] = (_Float16*)alloc((size_t)H_ * H_ * 2);
        WprA[l]  = (_Float16*)alloc((size_t)H_ * H_ * 2);
        WprB[l]  = (_Float16*)alloc((size_t)H_ * H_ * 2);
    }
    _Float16* Ws1rT = (_Float16*)alloc((size_t)H_ * H_ * 2);
    _Float16* Ws1tT = (_Float16*)alloc((size_t)H_ * H_ * 2);

    // ---- helpers -----------------------------------------------------------
    auto cvt = [&](const float* src, _Float16* dst, int n) {
        nav_cvt16_kernel<<<(n + 255) / 256, 256, 0, stream>>>(src, dst, n);
    };
    auto packwt = [&](const float* W, _Float16* dst, int K) {
        int n = K * H_;
        nav_packwt_kernel<<<(n + 255) / 256, 256, 0, stream>>>(W, dst, K, H_);
    };
    auto gemm = [&](const _Float16* A, const _Float16* Wt, const float* bias,
                    float* oF, _Float16* oH, int M, int K) {
        dim3 grid(M / 64, H_ / 64);
        nav_gemm_wmma_kernel<<<grid, 256, 0, stream>>>(A, Wt, bias, oF, oH, M, H_, K);
    };

    // ---- weight packing (f32 -> f16, transposed to [N][K]) -----------------
    packwt(Wp, WpT, D_);
    const size_t WSTRIDE = (size_t)2 * H_ * H_;     // per-layer (2H,H) stride
    const size_t HALF    = (size_t)H_ * H_;
    for (int l = 0; l < L_; ++l) {
        packwt(W_r2t + l * WSTRIDE,        Wr2tA[l], H_);
        packwt(W_r2t + l * WSTRIDE + HALF, Wr2tB[l], H_);
        packwt(W_pt  + l * WSTRIDE,        WptA[l],  H_);
        packwt(W_pt  + l * WSTRIDE + HALF, WptB[l],  H_);
        packwt(W_t2r + l * WSTRIDE,        Wt2rA[l], H_);
        packwt(W_t2r + l * WSTRIDE + HALF, Wt2rB[l], H_);
        packwt(W_pr  + l * WSTRIDE,        WprA[l],  H_);
        packwt(W_pr  + l * WSTRIDE + HALF, WprB[l],  H_);
    }
    packwt(Ws1,        Ws1rT, H_);
    packwt(Ws1 + HALF, Ws1tT, H_);

    // ---- inputs: target_raw mean + f16 conversion --------------------------
    nav_mean_i_kernel<<<(MS * D_ + 255) / 256, 256, 0, stream>>>(edge_raw, target_raw);
    cvt(robot_raw, A_rob16, MS * D_);
    cvt(target_raw, A_tgt16, MS * D_);
    cvt(edge_raw, A_edge16, ME * D_);

    // ---- projection --------------------------------------------------------
    gemm(A_rob16,  WpT, bp, robot_h,  robot_h16,  MS, D_);
    gemm(A_tgt16,  WpT, bp, target_h, target_h16, MS, D_);
    gemm(A_edge16, WpT, bp, nullptr,  edge_h16,   ME, D_);   // f16 only (GEMM-A reuse)

    // ---- message-passing layers -------------------------------------------
    for (int l = 0; l < L_; ++l) {
        // robot -> target
        gemm(robot_h16, Wr2tA[l], nullptr, u1, nullptr, MS, H_);        // rWa
        gemm(edge_h16,  Wr2tB[l], nullptr, eWb, nullptr, ME, H_);       // big
        nav_agg_kernel<<<(MS * H_ + 255) / 256, 256, 0, stream>>>(
            u1, eWb, b_r2t + l * H_, agg16, /*stepX=*/1, /*stepY=*/N_); // mean over i
        gemm(target_h16, WptA[l], nullptr, u1, nullptr, MS, H_);
        gemm(agg16,      WptB[l], nullptr, u2, nullptr, MS, H_);
        nav_ln_kernel<<<MS, 256, 0, stream>>>(target_h, u1, u2, b_pt + l * H_,
                                              g_t + l * H_, be_t + l * H_, target_h16);
        // target -> robot
        gemm(target_h16, Wt2rA[l], nullptr, u1, nullptr, MS, H_);       // tWa
        gemm(edge_h16,   Wt2rB[l], nullptr, eWb, nullptr, ME, H_);      // big
        nav_agg_kernel<<<(MS * H_ + 255) / 256, 256, 0, stream>>>(
            u1, eWb, b_t2r + l * H_, agg16, /*stepX=*/N_, /*stepY=*/1); // mean over j
        gemm(robot_h16, WprA[l], nullptr, u1, nullptr, MS, H_);
        gemm(agg16,     WprB[l], nullptr, u2, nullptr, MS, H_);
        nav_ln_kernel<<<MS, 256, 0, stream>>>(robot_h, u1, u2, b_pr + l * H_,
                                              g_r + l * H_, be_r + l * H_, robot_h16);
    }

    // ---- score head + Sinkhorn --------------------------------------------
    gemm(robot_h16,  Ws1rT, nullptr, u1, nullptr, MS, H_);   // rS
    gemm(target_h16, Ws1tT, nullptr, u2, nullptr, MS, H_);   // tS
    nav_score_kernel<<<ME, 256, 0, stream>>>(u1, u2, bs1, Ws2, bs2, out + ME);
    nav_sinkhorn_kernel<<<B_, N_, 0, stream>>>(out + ME, out);
}
